// stcModel_73126113182152
// MI455X (gfx1250) — matrix operand order repaired and checked by reading, not compiled, not run on hardware
//
#include <hip/hip_runtime.h>
#include <math.h>

// ---------------------------------------------------------------------------
// Shapes
// ---------------------------------------------------------------------------
#define BB 4
#define TT 300
#define SS 63
#define DD 512
#define LL 6
#define HH 8
#define MM (BB * TT * SS)   // 75600 real token rows (multiple of 16)
#define MP 75616            // GEMM row space padded to a multiple of 32
#define NW 8                // N sub-tiles per wave (16 cols each -> 128)

typedef __attribute__((ext_vector_type(16))) __bf16 v16bf;
typedef __attribute__((ext_vector_type(8)))  __bf16 v8bf;
typedef __attribute__((ext_vector_type(8)))  float  v8f;

__device__ __forceinline__ __bf16 tobf(float f) {
  // f32 -> bf16 round-to-nearest-even via bit ops
  unsigned u = __builtin_bit_cast(unsigned, f);
  unsigned r = (u + 0x7FFFu + ((u >> 16) & 1u)) >> 16;
  unsigned short s = (unsigned short)r;
  return __builtin_bit_cast(__bf16, s);
}

__device__ __forceinline__ float gelu_exact(float x) {
  return 0.5f * x * (1.0f + erff(x * 0.70710678118654752440f));
}

// ---------------------------------------------------------------------------
// Elementwise helpers
// ---------------------------------------------------------------------------
__global__ void k_zero(float* p, int n) {
  int i = blockIdx.x * blockDim.x + threadIdx.x;
  if (i < n) p[i] = 0.f;
}

__global__ void k_convert_w(const float* __restrict__ w, __bf16* __restrict__ o, int n) {
  int i = blockIdx.x * blockDim.x + threadIdx.x;
  if (i < n) o[i] = tobf(w[i]);
}

__global__ void k_add_pos(const float* __restrict__ xin, const float* __restrict__ spos,
                          const float* __restrict__ tpos, float* __restrict__ X) {
  long id = (long)blockIdx.x * blockDim.x + threadIdx.x;
  if (id >= (long)MM * DD) return;
  int d = (int)(id % DD);
  long r = id / DD;
  int s = (int)(r % SS);
  int t = (int)((r / SS) % TT);
  X[id] = xin[id] + spos[s * DD + d] + tpos[t * DD + d];
}

__global__ void k_pack_x(const float* __restrict__ X, __bf16* __restrict__ out) {
  long id = (long)blockIdx.x * blockDim.x + threadIdx.x;
  if (id >= (long)MM * DD) return;
  int d = (int)(id % DD);
  long m = id / DD;
  out[m * 1536 + d] = tobf(X[m * DD + d]);
}

// ---------------------------------------------------------------------------
// LayerNorm (f32 in -> bf16 out), one 128-thread block per token row
// ---------------------------------------------------------------------------
__global__ void k_layernorm(const float* __restrict__ x, const float* __restrict__ g,
                            const float* __restrict__ b, __bf16* __restrict__ out) {
  const int row = blockIdx.x;
  const int tid = threadIdx.x;    // 128
  const float* xr = x + (size_t)row * DD;
  float v[4];
  float s1 = 0.f, s2 = 0.f;
#pragma unroll
  for (int j = 0; j < 4; ++j) {
    v[j] = xr[tid + j * 128];
    s1 += v[j];
    s2 += v[j] * v[j];
  }
  __shared__ float r1[128], r2[128];
  r1[tid] = s1; r2[tid] = s2;
  __syncthreads();
  for (int off = 64; off > 0; off >>= 1) {
    if (tid < off) { r1[tid] += r1[tid + off]; r2[tid] += r2[tid + off]; }
    __syncthreads();
  }
  float mean = r1[0] * (1.f / 512.f);
  float var  = r2[0] * (1.f / 512.f) - mean * mean;
  float rstd = rsqrtf(var + 1e-5f);
  __bf16* o = out + (size_t)row * DD;
#pragma unroll
  for (int j = 0; j < 4; ++j) {
    int i = tid + j * 128;
    o[i] = tobf((v[j] - mean) * rstd * g[i] + b[i]);
  }
}

// ---------------------------------------------------------------------------
// WMMA bf16 GEMM: out[M,N] = act(A[M,K] @ W[K,N] + bias) (+ residual)
// Each wave32 computes a 32x128 strip: MW=2 M-tiles x NW=8 N-tiles of 16x16,
// K stepped by 32 with v_wmma_f32_16x16x32_bf16. Per k-step each lane loads
// 2 A fragments (2x16B each) + 8 B fragments (32B each) and issues 16 WMMAs
// -> 16 WMMAs per 320B/lane, ~3.2x the operand reuse of a 16x16-per-wave tiling.
// A fragment: lane l holds m = l%16, k-halves [kh..kh+7],[kh+16..kh+23],
//             kh = (l>=16)?8:0 -> two contiguous 16B loads.
// B fragment: lane l holds column n = l%16, k = kh+(s&7)+16*(s>>3)
//             -> strided gather of 16 elements (W row-major [K,N]).
// C/D: vgpr r, lane l -> (m = r + (l>=16?8:0), n = l%16).
// ---------------------------------------------------------------------------
template <bool GELU, bool BF16OUT, bool BIAS, bool RES>
__global__ __launch_bounds__(128) void k_gemm(const __bf16* __restrict__ A,
                                              const __bf16* __restrict__ W,
                                              const float* __restrict__ bias,
                                              const float* __restrict__ res,
                                              void* __restrict__ outp,
                                              int Mr, int N, int K) {
  const int lane = threadIdx.x & 31;
  const int wave = threadIdx.x >> 5;
  const int Nstrips = N >> 7;                                 // N / 128
  const long strip  = (long)blockIdx.x * 4 + wave;            // wave-uniform
  const long total  = (long)(Mr >> 5) * Nstrips;
  if (strip >= total) return;                                 // EXEC stays all-1s per wave
  const int mt = (int)(strip / Nstrips);
  const int ns = (int)(strip % Nstrips);
  const int row0 = mt << 5;
  const int col0 = ns << 7;
  const int mloc  = lane & 15;
  const int khalf = (lane >> 4) << 3;
  const __bf16* arow0 = A + (size_t)(row0 + mloc) * K + khalf;
  const __bf16* arow1 = arow0 + (size_t)16 * K;
  const __bf16* brow  = W + (size_t)khalf * N + col0 + mloc;

  v8f acc[2][NW];
#pragma unroll
  for (int mw = 0; mw < 2; ++mw)
#pragma unroll
    for (int nw = 0; nw < NW; ++nw)
      acc[mw][nw] = (v8f){0.f, 0.f, 0.f, 0.f, 0.f, 0.f, 0.f, 0.f};

  for (int kk = 0; kk < K; kk += 32) {
    union { v16bf v; v8bf h[2]; } a0, a1;
    a0.h[0] = *(const v8bf*)(arow0 + kk);
    a0.h[1] = *(const v8bf*)(arow0 + kk + 16);
    a1.h[0] = *(const v8bf*)(arow1 + kk);
    a1.h[1] = *(const v8bf*)(arow1 + kk + 16);
    __builtin_prefetch(arow0 + kk + 32, 0, 0);                // global_prefetch_b8
    __builtin_prefetch(arow1 + kk + 32, 0, 0);
    const __bf16* bk = brow + (size_t)kk * N;
    __builtin_prefetch(bk + (size_t)32 * N, 0, 0);
#pragma unroll
    for (int nw = 0; nw < NW; ++nw) {
      v16bf bfr;
#pragma unroll
      for (int s = 0; s < 16; ++s)
        bfr[s] = bk[(size_t)((s & 7) + ((s >> 3) << 4)) * N + (nw << 4)];
      acc[0][nw] = __builtin_amdgcn_wmma_f32_16x16x32_bf16(false, a0.v, false, bfr,
                                                           (short)0, acc[0][nw], false, false);
      acc[1][nw] = __builtin_amdgcn_wmma_f32_16x16x32_bf16(false, a1.v, false, bfr,
                                                           (short)0, acc[1][nw], false, false);
    }
  }

#pragma unroll
  for (int mw = 0; mw < 2; ++mw) {
    const int mbase = row0 + (mw << 4) + ((lane >> 4) << 3);
#pragma unroll
    for (int nw = 0; nw < NW; ++nw) {
      const int n = col0 + (nw << 4) + (lane & 15);
      float bv = 0.f;
      if (BIAS) bv = bias[n];
#pragma unroll
      for (int r = 0; r < 8; ++r) {
        float v = acc[mw][nw][r] + bv;
        if (GELU) v = gelu_exact(v);
        size_t idx = (size_t)(mbase + r) * N + n;
        if (RES) v += res[idx];
        if (BF16OUT) ((__bf16*)outp)[idx] = tobf(v);
        else         ((float*)outp)[idx]  = v;
      }
    }
  }
}

// ---------------------------------------------------------------------------
// Attention. qkv layout: token row of 1536 = (channel c, j) -> c*3 + j,
// j = 0/1/2 for q/k/v. Spatial heads use c in [0,256), temporal c in [256,512).
// Output channel layout (concat then reshape): spatial -> h*64 + d,
// temporal -> h*64 + 32 + d.  scale = 256^-0.5 = 1/16.
// Online softmax, K/V staged in LDS.
// ---------------------------------------------------------------------------
__global__ void k_attn_spatial(const float* __restrict__ qkv, __bf16* __restrict__ out) {
  const int bid = blockIdx.x;   // B*T*H
  const int h  = bid & 7;
  const int bt = bid >> 3;
  extern __shared__ float sm[];
  float* Ksh = sm;              // [63][32]
  float* Vsh = sm + SS * 32;
  const int tid = threadIdx.x;  // 64
  const size_t rowbase = (size_t)bt * SS * 1536;
  const int ch = h * 32;
  if (tid < SS) {
    const float* p = qkv + rowbase + (size_t)tid * 1536;
    for (int d = 0; d < 32; ++d) {
      Ksh[tid * 32 + d] = p[(ch + d) * 3 + 1];
      Vsh[tid * 32 + d] = p[(ch + d) * 3 + 2];
    }
  }
  __syncthreads();
  if (tid >= SS) return;
  const float* p = qkv + rowbase + (size_t)tid * 1536;
  float q[32];
#pragma unroll
  for (int d = 0; d < 32; ++d) q[d] = p[(ch + d) * 3];
  float mx = -1e30f, l = 0.f;
  float acc[32];
#pragma unroll
  for (int d = 0; d < 32; ++d) acc[d] = 0.f;
  for (int j = 0; j < SS; ++j) {
    float s = 0.f;
#pragma unroll
    for (int d = 0; d < 32; ++d) s += q[d] * Ksh[j * 32 + d];
    s *= 0.0625f;
    float mn = fmaxf(mx, s);
    float corr = __expf(mx - mn);
    float pj   = __expf(s - mn);
    l = l * corr + pj;
#pragma unroll
    for (int d = 0; d < 32; ++d) acc[d] = acc[d] * corr + pj * Vsh[j * 32 + d];
    mx = mn;
  }
  float inv = 1.f / l;
  __bf16* o = out + ((size_t)bt * SS + tid) * 512 + h * 64;
#pragma unroll
  for (int d = 0; d < 32; ++d) o[d] = tobf(acc[d] * inv);
}

__global__ void k_attn_temporal(const float* __restrict__ qkv, __bf16* __restrict__ out) {
  const int bid = blockIdx.x;   // B*S*H
  const int h  = bid & 7;
  const int bs = bid >> 3;
  const int b = bs / SS;
  const int s = bs % SS;
  extern __shared__ float sm[];
  float* Ksh = sm;              // [300][32]
  float* Vsh = sm + TT * 32;
  const int tid = threadIdx.x;  // 320
  const int ch = 256 + h * 32;
  if (tid < TT) {
    const size_t m = ((size_t)b * TT + tid) * SS + s;
    const float* p = qkv + m * 1536;
    for (int d = 0; d < 32; ++d) {
      Ksh[tid * 32 + d] = p[(ch + d) * 3 + 1];
      Vsh[tid * 32 + d] = p[(ch + d) * 3 + 2];
    }
  }
  __syncthreads();
  if (tid >= TT) return;
  const size_t m = ((size_t)b * TT + tid) * SS + s;
  const float* p = qkv + m * 1536;
  float q[32];
#pragma unroll
  for (int d = 0; d < 32; ++d) q[d] = p[(ch + d) * 3];
  float mx = -1e30f, l = 0.f;
  float acc[32];
#pragma unroll
  for (int d = 0; d < 32; ++d) acc[d] = 0.f;
  for (int j = 0; j < TT; ++j) {
    float sc = 0.f;
#pragma unroll
    for (int d = 0; d < 32; ++d) sc += q[d] * Ksh[j * 32 + d];
    sc *= 0.0625f;
    float mn = fmaxf(mx, sc);
    float corr = __expf(mx - mn);
    float pj   = __expf(sc - mn);
    l = l * corr + pj;
#pragma unroll
    for (int d = 0; d < 32; ++d) acc[d] = acc[d] * corr + pj * Vsh[j * 32 + d];
    mx = mn;
  }
  float inv = 1.f / l;
  __bf16* o = out + m * 512 + h * 64 + 32;
#pragma unroll
  for (int d = 0; d < 32; ++d) o[d] = tobf(acc[d] * inv);
}

// ---------------------------------------------------------------------------
// Resample: real(fft)->truncate->real(ifft)->linear interp back to T folds to a
// single [T,T] mixing matrix  RM[t,tt] = (1-w)M(i0,tt) + wM(i1,tt),
// M(tp,tt) = (300*Lf)^-1/2 * sum_k cos(2pi k tt/300) cos(2pi k tp/Lf).
// ---------------------------------------------------------------------------
__global__ void k_build_rm(float* __restrict__ RM, int Lf) {
  int id = blockIdx.x * blockDim.x + threadIdx.x;
  if (id >= TT * TT) return;
  int tt = id % TT;
  int t  = id / TT;
  float src = ((float)t + 0.5f) * ((float)Lf / (float)TT) - 0.5f;
  src = fminf(fmaxf(src, 0.f), (float)(Lf - 1));
  int i0 = (int)floorf(src);
  int i1 = min(i0 + 1, Lf - 1);
  float w = src - (float)i0;
  const float c1 = 6.28318530717958647692f / (float)TT;
  const float c2 = 6.28318530717958647692f / (float)Lf;
  float s0 = 0.f, s1 = 0.f;
  for (int k = 0; k < Lf; ++k) {
    float ck = cosf(c1 * (float)k * (float)tt);
    s0 += ck * cosf(c2 * (float)k * (float)i0);
    s1 += ck * cosf(c2 * (float)k * (float)i1);
  }
  float invs = rsqrtf((float)TT * (float)Lf);
  RM[id] = invs * ((1.f - w) * s0 + w * s1);
}

__global__ void k_resample(const float* __restrict__ X, const float* __restrict__ RM,
                           __bf16* __restrict__ out, int coff) {
  long id = (long)blockIdx.x * blockDim.x + threadIdx.x;
  if (id >= (long)MM * DD) return;
  int d = (int)(id % DD);
  long r = id / DD;                  // token row ((b*T+t)*S + s)
  int s = (int)(r % SS);
  long bt = r / SS;
  int t = (int)(bt % TT);
  int b = (int)(bt / TT);
  const float* xp = X + (((size_t)b * TT) * SS + s) * DD + d;   // tt = 0
  const float* rm = RM + (size_t)t * TT;
  float acc = 0.f;
  for (int tt2 = 0; tt2 < TT; ++tt2)
    acc += rm[tt2] * xp[(size_t)tt2 * SS * DD];
  out[r * 1536 + coff + d] = tobf(acc);
}

// ---------------------------------------------------------------------------
// Head: LN(y) @ head_w + head_b, then mean over s via atomicAdd per (b,t)
// ---------------------------------------------------------------------------
__global__ void k_head(const float* __restrict__ y, const float* __restrict__ g,
                       const float* __restrict__ b, const float* __restrict__ hw,
                       const float* __restrict__ hb, float* __restrict__ out) {
  const int row = blockIdx.x;
  const int tid = threadIdx.x;    // 128
  const float* yr = y + (size_t)row * DD;
  float v[4];
  float s1 = 0.f, s2 = 0.f;
#pragma unroll
  for (int j = 0; j < 4; ++j) {
    v[j] = yr[tid + j * 128];
    s1 += v[j];
    s2 += v[j] * v[j];
  }
  __shared__ float r1[128], r2[128];
  r1[tid] = s1; r2[tid] = s2;
  __syncthreads();
  for (int off = 64; off > 0; off >>= 1) {
    if (tid < off) { r1[tid] += r1[tid + off]; r2[tid] += r2[tid + off]; }
    __syncthreads();
  }
  float mean = r1[0] * (1.f / 512.f);
  float var  = r2[0] * (1.f / 512.f) - mean * mean;
  float rstd = rsqrtf(var + 1e-5f);
  float dotp = 0.f;
#pragma unroll
  for (int j = 0; j < 4; ++j) {
    int i = tid + j * 128;
    dotp += ((v[j] - mean) * rstd * g[i] + b[i]) * hw[i];
  }
  __syncthreads();
  r1[tid] = dotp;
  __syncthreads();
  for (int off = 64; off > 0; off >>= 1) {
    if (tid < off) r1[tid] += r1[tid + off];
    __syncthreads();
  }
  if (tid == 0)
    atomicAdd(&out[row / SS], (r1[0] + hb[0]) * (1.f / 63.f));
}

// ---------------------------------------------------------------------------
// Host-side GEMM launcher (Mr must be a multiple of 32, N of 128, K of 32)
// ---------------------------------------------------------------------------
template <bool GELU, bool BF16OUT, bool BIAS, bool RES>
static void gemm(const __bf16* A, const __bf16* W, const float* bias, const float* res,
                 void* out, int Mr, int N, int K, hipStream_t st) {
  long strips = (long)(Mr / 32) * (N / 128);
  int blocks = (int)((strips + 3) / 4);          // 4 waves (32x128 strips) per block
  k_gemm<GELU, BF16OUT, BIAS, RES><<<blocks, 128, 0, st>>>(A, W, bias, res, out, Mr, N, K);
}

// ---------------------------------------------------------------------------
// Workspace layout (bytes); ~930 MB total, regions reused across stages.
// All GEMM operand buffers are sized with MP (padded) rows; pad rows carry
// harmless garbage and their outputs are never consumed.
// ---------------------------------------------------------------------------
static const size_t OFF_A    = (size_t)MP * DD * 4;                 // after X (f32 MP*512)
static const size_t OFF_Q    = OFF_A + (size_t)MP * 1536 * 2;       // after ABUF (bf16 MP*1536)
static const size_t OFF_D    = OFF_Q + (size_t)MP * 1536 * 4;       // after QKV (f32 MP*1536)
static const size_t OFF_W    = OFF_D + (size_t)MP * DD * 2;         // after DBUF (bf16 MP*512)
static const size_t OFF_R150 = OFF_W + (size_t)1536 * 1536 * 2;     // after WBUF
static const size_t OFF_R75  = OFF_R150 + (size_t)TT * TT * 4;

extern "C" void kernel_launch(void* const* d_in, const int* in_sizes, int n_in,
                              void* d_out, int out_size, void* d_ws, size_t ws_size,
                              hipStream_t stream) {
  (void)in_sizes; (void)n_in; (void)out_size; (void)ws_size;
  const float* x_in   = (const float*)d_in[0];
  const float* spos   = (const float*)d_in[1];
  const float* tpos   = (const float*)d_in[2];
  const float* ln1_g  = (const float*)d_in[3];
  const float* ln1_b  = (const float*)d_in[4];
  const float* qkv_w  = (const float*)d_in[5];
  const float* qkv_b  = (const float*)d_in[6];
  const float* proj_w = (const float*)d_in[7];
  const float* proj_b = (const float*)d_in[8];
  const float* ln2_g  = (const float*)d_in[9];
  const float* ln2_b  = (const float*)d_in[10];
  const float* fc1_w  = (const float*)d_in[11];
  const float* fc2_w  = (const float*)d_in[12];
  const float* ff1_w  = (const float*)d_in[13];
  const float* ff1_b  = (const float*)d_in[14];
  const float* ff2_w  = (const float*)d_in[15];
  const float* ff2_b  = (const float*)d_in[16];
  const float* hln_g  = (const float*)d_in[17];
  const float* hln_b  = (const float*)d_in[18];
  const float* head_w = (const float*)d_in[19];
  const float* head_b = (const float*)d_in[20];

  char* ws = (char*)d_ws;
  float*  X     = (float*)ws;                     // residual stream (also ff2 output)
  __bf16* ABUF  = (__bf16*)(ws + OFF_A);          // LN out / attn out / x300
  float*  QKV   = (float*)(ws + OFF_Q);           // qkv f32
  __bf16* HBUF  = (__bf16*)(ws + OFF_Q);          // ff1 output aliases QKV region
  __bf16* DBUF  = (__bf16*)(ws + OFF_D);          // fc1 gelu output
  __bf16* WBUF  = (__bf16*)(ws + OFF_W);          // staged bf16 weights
  float*  RM150 = (float*)(ws + OFF_R150);
  float*  RM75  = (float*)(ws + OFF_R75);

  const long NE = (long)MM * DD;                  // 38,707,200
  const int TPB = 256;
  const int nblk_e = (int)((NE + TPB - 1) / TPB);

  k_zero<<<(1200 + 255) / 256, 256, 0, stream>>>((float*)d_out, BB * TT);
  k_add_pos<<<nblk_e, TPB, 0, stream>>>(x_in, spos, tpos, X);
  k_build_rm<<<(TT * TT + 255) / 256, 256, 0, stream>>>(RM150, 150);
  k_build_rm<<<(TT * TT + 255) / 256, 256, 0, stream>>>(RM75, 75);

  for (int i = 0; i < LL; ++i) {
    // attention block
    k_layernorm<<<MM, 128, 0, stream>>>(X, ln1_g + i * DD, ln1_b + i * DD, ABUF);
    k_convert_w<<<(512 * 1536 + 255) / 256, 256, 0, stream>>>(
        qkv_w + (size_t)i * 512 * 1536, WBUF, 512 * 1536);
    gemm<false, false, true, false>(ABUF, WBUF, qkv_b + i * 1536, nullptr, QKV,
                                    MP, 1536, 512, stream);
    k_attn_spatial<<<BB * TT * HH, 64, SS * 32 * 2 * sizeof(float), stream>>>(QKV, ABUF);
    k_attn_temporal<<<BB * SS * HH, 320, TT * 32 * 2 * sizeof(float), stream>>>(QKV, ABUF);
    k_convert_w<<<(512 * 512 + 255) / 256, 256, 0, stream>>>(
        proj_w + (size_t)i * 512 * 512, WBUF, 512 * 512);
    gemm<false, false, true, true>(ABUF, WBUF, proj_b + i * DD, X, X, MP, 512, 512, stream);
    // MLP block (no biases on fc1/fc2)
    k_layernorm<<<MM, 128, 0, stream>>>(X, ln2_g + i * DD, ln2_b + i * DD, ABUF);
    k_convert_w<<<(512 * 512 + 255) / 256, 256, 0, stream>>>(
        fc1_w + (size_t)i * 512 * 512, WBUF, 512 * 512);
    gemm<true, true, false, false>(ABUF, WBUF, nullptr, nullptr, DBUF, MP, 512, 512, stream);
    k_convert_w<<<(512 * 512 + 255) / 256, 256, 0, stream>>>(
        fc2_w + (size_t)i * 512 * 512, WBUF, 512 * 512);
    gemm<false, false, false, true>(DBUF, WBUF, nullptr, X, X, MP, 512, 512, stream);
  }

  // x300 = concat([x, resample150, resample75]) as bf16 GEMM operand
  k_pack_x<<<nblk_e, TPB, 0, stream>>>(X, ABUF);
  k_resample<<<nblk_e, TPB, 0, stream>>>(X, RM150, ABUF, 512);
  k_resample<<<nblk_e, TPB, 0, stream>>>(X, RM75, ABUF, 1024);

  k_convert_w<<<(1536 * 1536 + 255) / 256, 256, 0, stream>>>(ff1_w, WBUF, 1536 * 1536);
  gemm<true, true, true, false>(ABUF, WBUF, ff1_b, nullptr, HBUF, MP, 1536, 1536, stream);
  k_convert_w<<<(1536 * 512 + 255) / 256, 256, 0, stream>>>(ff2_w, WBUF, 1536 * 512);
  gemm<false, false, true, false>(HBUF, WBUF, ff2_b, nullptr, X, MP, 512, 1536, stream);

  k_head<<<MM, 128, 0, stream>>>(X, hln_g, hln_b, head_w, head_b, (float*)d_out);
}
